// Method_RNN_Generation_35502199669482
// MI455X (gfx1250) — matrix operand order repaired and checked by reading, not compiled, not run on hardware
//
#include <hip/hip_runtime.h>
#include <hip/hip_bf16.h>

typedef __attribute__((ext_vector_type(16))) _Float16 v16h;
typedef __attribute__((ext_vector_type(8)))  _Float16 v8h;
typedef __attribute__((ext_vector_type(8)))  float    v8f;
typedef __attribute__((ext_vector_type(4)))  int      v4i;

#define WMMA_F16(a, b, c) \
  __builtin_amdgcn_wmma_f32_16x16x32_f16(false, (a), false, (b), (short)0, (c), false, false)

// ---- CDNA5 async global->LDS copy (ASYNCcnt path), guarded ----------------
#if defined(__has_builtin)
#if __has_builtin(__builtin_amdgcn_global_load_async_to_lds_b128) && \
    __has_builtin(__builtin_amdgcn_s_wait_asynccnt)
#define HAVE_ASYNC 1
#endif
#endif
#ifndef HAVE_ASYNC
#define HAVE_ASYNC 0
#endif

#if HAVE_ASYNC
typedef __attribute__((address_space(1))) v4i* as1_v4i_p;  // global int4*
typedef __attribute__((address_space(3))) v4i* as3_v4i_p;  // LDS int4*
__device__ __forceinline__ void async_cp_b128(const void* gsrc, void* ldst) {
  __builtin_amdgcn_global_load_async_to_lds_b128((as1_v4i_p)gsrc, (as3_v4i_p)ldst, 0, 0);
}
#endif

// ---------------------------------------------------------------------------
// Fragment loaders for V_WMMA_F32_16X16X32_F16 (wave32 layouts per ISA 7.12.2)
// A (16x32 f16, row-major source [M][128]):
//   lane L: row = m0 + L%16 ; elems 0..7  -> K = k0 + (L<16?0:8) + j
//                             elems 8..15 -> same + 16
// B (32x16 f16, source stored row-major [N][128] i.e. W[n][k]):
//   lane L: col = n0 + L%16 ; elems 0..7 -> K = k0 + (L<16?0:8) + j ; elems 8..15 -> same + 16
// C/D (16x16 f32): lane L: col = n0 + L%16 ; elem r -> row = m0 + (L<16?r:8+r)
// ---------------------------------------------------------------------------
__device__ __forceinline__ v16h load_a_frag(const _Float16* A, int m0, int k0) {
  const int l = threadIdx.x & 31;
  const _Float16* p = A + (((size_t)(m0 + (l & 15))) << 7) + k0 + ((l >> 4) << 3);
  v8h lo = *(const v8h*)p;
  v8h hi = *(const v8h*)(p + 16);
  v16h r;
#pragma unroll
  for (int j = 0; j < 8; ++j) { r[j] = lo[j]; r[j + 8] = hi[j]; }
  return r;
}

__device__ __forceinline__ v16h load_b_frag(const _Float16* W, int n0, int k0) {
  const int l = threadIdx.x & 31;
  const _Float16* p = W + (((size_t)(n0 + (l & 15))) << 7) + k0 + ((l >> 4) << 3);
  v8h lo = *(const v8h*)p;
  v8h hi = *(const v8h*)(p + 16);
  v16h r;
#pragma unroll
  for (int j = 0; j < 8; ++j) { r[j] = lo[j]; r[j + 8] = hi[j]; }
  return r;
}

// ---------------------------------------------------------------------------
// One-shot weight pack: f32 [N][K] -> f16 [N][128], zero-padded
// ---------------------------------------------------------------------------
__global__ void pack_w_f16(const float* __restrict__ W, _Float16* __restrict__ Wp,
                           int N, int K) {
  const int idx = blockIdx.x * 256 + threadIdx.x;
  if (idx >= (N << 7)) return;
  const int n = idx >> 7, k = idx & 127;
  Wp[idx] = (k < K) ? (_Float16)W[n * K + k] : (_Float16)0.f;
}

// ---------------------------------------------------------------------------
// Embedding gather: xi[s*B+b][k] = emb[text[b][s]][k] (f16, K padded 100->128)
// ---------------------------------------------------------------------------
__global__ void embed_gather(const int* __restrict__ text,
                             const float* __restrict__ emb,
                             _Float16* __restrict__ xi) {
  size_t idx = (size_t)blockIdx.x * blockDim.x + threadIdx.x;
  if (idx >= ((size_t)262144 << 7)) return;
  const int    k  = (int)(idx & 127);
  const size_t sb = idx >> 7;
  const int s = (int)(sb >> 8);
  const int b = (int)(sb & 255);
  const int tok = text[b * 1024 + s];
  xi[idx] = (k < 100) ? (_Float16)emb[tok * 100 + k] : (_Float16)0.f;
}

// ---------------------------------------------------------------------------
// Batch-parallel WMMA GEMM, K=128 fixed: Out[M][NTOT] = A[M][128] * Wp[N][128]^T
// Block: 256 threads, 64-row strip. A strip staged in LDS once (16 KB, async
// when available), read by all 8 waves; B fragments hoisted to VGPRs;
// compute is ds_load + 16 WMMAs per wave.
// ---------------------------------------------------------------------------
__global__ __launch_bounds__(256) void wmma_gemm_f16(
    const _Float16* __restrict__ A,
    const _Float16* __restrict__ Wp,
    int nTiles, int NTOT,
    _Float16* __restrict__ outH, float* __restrict__ outF,
    const float* __restrict__ bias) {
  __shared__ _Float16 As[64 * 128];                 // 16 KB
  const int m0 = blockIdx.x << 6;

  // Stage A strip: 8192 halfs = 1024 x 16B chunks, 4 per thread
#if HAVE_ASYNC
#pragma unroll
  for (int i = 0; i < 4; ++i) {
    const int c = threadIdx.x + (i << 8);
    const int row = c >> 4, kk = (c & 15) << 3;
    async_cp_b128(&A[(((size_t)(m0 + row)) << 7) + kk], &As[(row << 7) + kk]);
  }
  __builtin_amdgcn_s_wait_asynccnt(0);
#else
#pragma unroll
  for (int c = threadIdx.x; c < 1024; c += 256) {
    const int row = c >> 4, kk = (c & 15) << 3;
    *(v8h*)&As[(row << 7) + kk] = *(const v8h*)&A[(((size_t)(m0 + row)) << 7) + kk];
  }
#endif
  __syncthreads();  // before wave-exit guard: all 256 threads participate

  const int wv = threadIdx.x >> 5;
  const int l  = threadIdx.x & 31;
  const int nt = blockIdx.y * 8 + wv;
  if (nt >= nTiles) return;                 // wave-uniform, EXEC stays full
  const int n0 = nt << 4;
  const int n  = n0 + (l & 15);

  // Hoist all B fragments for this n-tile (8 x global_load_b128, one wait)
  v16h bf[4];
#pragma unroll
  for (int kt = 0; kt < 4; ++kt) bf[kt] = load_b_frag(Wp, n0, kt << 5);

  v8f acc[4];
#pragma unroll
  for (int mt = 0; mt < 4; ++mt)
#pragma unroll
    for (int r = 0; r < 8; ++r) acc[mt][r] = 0.f;

#pragma unroll
  for (int kt = 0; kt < 4; ++kt)
#pragma unroll
    for (int mt = 0; mt < 4; ++mt) {
      v16h a = load_a_frag(As, mt << 4, kt << 5);
      acc[mt] = WMMA_F16(a, bf[kt], acc[mt]);
    }

  const int rbase = (l >> 4) << 3;
  if (outH) {
#pragma unroll
    for (int mt = 0; mt < 4; ++mt)
#pragma unroll
      for (int r = 0; r < 8; ++r)
        outH[(size_t)(m0 + (mt << 4) + rbase + r) * NTOT + n] = (_Float16)acc[mt][r];
  } else {
    const float bb = bias ? bias[n] : 0.f;
#pragma unroll
    for (int mt = 0; mt < 4; ++mt)
#pragma unroll
      for (int r = 0; r < 8; ++r)
        outF[(size_t)(m0 + (mt << 4) + rbase + r) * NTOT + n] = acc[mt][r] + bb;
  }
}

// ---------------------------------------------------------------------------
// Persistent LSTM recurrence. One block = 16 batch rows for all 1024 steps.
// 8 waves; wave jb owns h-columns [jb*16, jb*16+16) and computes the four
// gate tiles (i,f,g,o) for that slice -> gate math stays in registers.
// Whh fragments hoisted into VGPRs (step-invariant). h exchanged via 4KB LDS.
// xg_t tiles (16x512 f16 = 16KB) are ping-pong staged into LDS via the async
// engine one step ahead, hiding global latency behind the step's WMMAs.
// ---------------------------------------------------------------------------
__global__ __launch_bounds__(256) void lstm_rec(
    const _Float16* __restrict__ xg,    // [S*B][512] f16 precomputed input proj
    const _Float16* __restrict__ Whhp,  // [512][128] f16 packed
    const float* __restrict__ bih, const float* __restrict__ bhh,
    _Float16* __restrict__ hs) {        // [S*B][128] f16 layer output
  __shared__ _Float16 hbuf[16 * 128];   // 4 KB h exchange
#if HAVE_ASYNC
  __shared__ _Float16 xbuf[2][16 * 512];  // 32 KB xg ping-pong
#endif
  const int l     = threadIdx.x & 31;
  const int jb    = threadIdx.x >> 5;          // 0..7
  const int b0    = blockIdx.x << 4;           // batch slice base
  const int col   = (jb << 4) + (l & 15);      // 0..127
  const int rbase = (l >> 4) << 3;             // 0 or 8

  // Hoist recurrent-weight B fragments: Bf[gate][kt]
  v16h Bf[4][4];
#pragma unroll
  for (int g = 0; g < 4; ++g)
#pragma unroll
    for (int kt = 0; kt < 4; ++kt)
      Bf[g][kt] = load_b_frag(Whhp, g * 128 + (jb << 4), kt << 5);

  float bias[4];
#pragma unroll
  for (int g = 0; g < 4; ++g) bias[g] = bih[g * 128 + col] + bhh[g * 128 + col];

  float c[8];
#pragma unroll
  for (int r = 0; r < 8; ++r) c[r] = 0.f;
  for (int i = threadIdx.x; i < 16 * 128; i += 256) hbuf[i] = (_Float16)0.f;

#if HAVE_ASYNC
  // Prologue: stage xg tile for s=0 into buffer 0 (4 async b128 per thread)
#pragma unroll
  for (int i = 0; i < 4; ++i) {
    const int ch = threadIdx.x + (i << 8);      // 0..1023
    const int row = ch >> 6, off = (ch & 63) << 3;
    async_cp_b128(&xg[(size_t)(b0 + row) * 512 + off], &xbuf[0][(row << 9) + off]);
  }
#endif
  __syncthreads();

  for (int s = 0; s < 1024; ++s) {
#if HAVE_ASYNC
    const int cur = s & 1;
    if (s + 1 < 1024) {   // stage next step's xg tile into the other buffer
#pragma unroll
      for (int i = 0; i < 4; ++i) {
        const int ch = threadIdx.x + (i << 8);
        const int row = ch >> 6, off = (ch & 63) << 3;
        async_cp_b128(&xg[(size_t)((s + 1) * 256 + b0 + row) * 512 + off],
                      &xbuf[cur ^ 1][(row << 9) + off]);
      }
    }
#else
    if (s + 1 < 1024) {
#pragma unroll
      for (int g = 0; g < 4; ++g)
        __builtin_prefetch(&xg[(size_t)((s + 1) * 256 + b0 + rbase) * 512 + g * 128 + col], 0, 3);
    }
#endif
    // A fragments of h_{t-1} from LDS
    v16h Af[4];
#pragma unroll
    for (int kt = 0; kt < 4; ++kt) Af[kt] = load_a_frag(hbuf, 0, kt << 5);
#if HAVE_ASYNC
    // step-s async batch done (in-order per wave: <=4 leaves only s+1's ops)
    if (s != 1023) __builtin_amdgcn_s_wait_asynccnt(4);
    else           __builtin_amdgcn_s_wait_asynccnt(0);
#endif
    __syncthreads();  // h reads done + step-s xg tile visible from all waves

    // acc = xg_t + (bih + bhh)
    v8f acc[4];
#pragma unroll
    for (int g = 0; g < 4; ++g) {
#if HAVE_ASYNC
#pragma unroll
      for (int r = 0; r < 8; ++r)
        acc[g][r] = (float)xbuf[cur][((rbase + r) << 9) + g * 128 + col] + bias[g];
#else
      const _Float16* xp = &xg[(size_t)(s * 256 + b0 + rbase) * 512 + g * 128 + col];
#pragma unroll
      for (int r = 0; r < 8; ++r) acc[g][r] = (float)xp[(size_t)r * 512] + bias[g];
#endif
    }
    // acc += h_{t-1} @ Whh^T   (16 WMMAs per wave per step)
#pragma unroll
    for (int g = 0; g < 4; ++g)
#pragma unroll
      for (int kt = 0; kt < 4; ++kt)
        acc[g] = WMMA_F16(Af[kt], Bf[g][kt], acc[g]);

    // Gates + cell update, fully in registers (identical lane mapping)
#pragma unroll
    for (int r = 0; r < 8; ++r) {
      const float gi = 1.f / (1.f + __expf(-acc[0][r]));
      const float gf = 1.f / (1.f + __expf(-acc[1][r]));
      const float gg = tanhf(acc[2][r]);
      const float go = 1.f / (1.f + __expf(-acc[3][r]));
      c[r] = gf * c[r] + gi * gg;
      const float h = go * tanhf(c[r]);
      const int row = rbase + r;
      hbuf[row * 128 + col] = (_Float16)h;
      hs[(size_t)(s * 256 + b0 + row) * 128 + col] = (_Float16)h;
    }
    __syncthreads();  // h_t visible to all waves before next step's A loads
  }
}

// ---------------------------------------------------------------------------
extern "C" void kernel_launch(void* const* d_in, const int* in_sizes, int n_in,
                              void* d_out, int out_size, void* d_ws, size_t ws_size,
                              hipStream_t stream) {
  const int*   text = (const int*)d_in[0];
  const float* emb  = (const float*)d_in[1];
  const float* fcW  = (const float*)d_in[2];
  const float* fcb  = (const float*)d_in[3];
  const float* Wih[3] = {(const float*)d_in[4], (const float*)d_in[8],  (const float*)d_in[12]};
  const float* Whh[3] = {(const float*)d_in[5], (const float*)d_in[9],  (const float*)d_in[13]};
  const float* bih[3] = {(const float*)d_in[6], (const float*)d_in[10], (const float*)d_in[14]};
  const float* bhh[3] = {(const float*)d_in[7], (const float*)d_in[11], (const float*)d_in[15]};

  // Workspace layout (bytes), all K padded to 128:
  //   xi  @ 0         : [262144][128] f16 =  67,108,864
  //   xg  @ 67108864  : [262144][512] f16 = 268,435,456
  //   hsA @ 335544320 : [262144][128] f16 =  67,108,864
  //   hsB @ 402653184 : [262144][128] f16 =  67,108,864
  //   packed weights @ 469762048 (f16 [N][128])
  char* ws = (char*)d_ws;
  _Float16* xi  = (_Float16*)(ws);
  _Float16* xg  = (_Float16*)(ws + (size_t)67108864);
  _Float16* hsA = (_Float16*)(ws + (size_t)335544320);
  _Float16* hsB = (_Float16*)(ws + (size_t)402653184);
  _Float16* Wih0p = (_Float16*)(ws + (size_t)469762048);  // 512*128*2 = 131072
  _Float16* Wih1p = (_Float16*)(ws + (size_t)469893120);
  _Float16* Wih2p = (_Float16*)(ws + (size_t)470024192);
  _Float16* fcWp  = (_Float16*)(ws + (size_t)470155264);  // 288*128*2 = 73728
  _Float16* Whh0p = (_Float16*)(ws + (size_t)470228992);
  _Float16* Whh1p = (_Float16*)(ws + (size_t)470360064);
  _Float16* Whh2p = (_Float16*)(ws + (size_t)470491136);  // end 470622208

  // Pack weights to f16 [N][128] (one-shot, tiny)
  pack_w_f16<<<256, 256, 0, stream>>>(Wih[0], Wih0p, 512, 100);
  pack_w_f16<<<256, 256, 0, stream>>>(Wih[1], Wih1p, 512, 128);
  pack_w_f16<<<256, 256, 0, stream>>>(Wih[2], Wih2p, 512, 128);
  pack_w_f16<<<144, 256, 0, stream>>>(fcW,    fcWp,  288, 128);
  pack_w_f16<<<256, 256, 0, stream>>>(Whh[0], Whh0p, 512, 128);
  pack_w_f16<<<256, 256, 0, stream>>>(Whh[1], Whh1p, 512, 128);
  pack_w_f16<<<256, 256, 0, stream>>>(Whh[2], Whh2p, 512, 128);

  embed_gather<<<131072, 256, 0, stream>>>(text, emb, xi);

  // Layer 0: xg = xi @ Wih0^T ; recurrence -> hsA
  wmma_gemm_f16<<<dim3(4096, 4), 256, 0, stream>>>(xi, Wih0p, 32, 512,
                                                   xg, nullptr, nullptr);
  lstm_rec<<<16, 256, 0, stream>>>(xg, Whh0p, bih[0], bhh[0], hsA);

  // Layer 1
  wmma_gemm_f16<<<dim3(4096, 4), 256, 0, stream>>>(hsA, Wih1p, 32, 512,
                                                   xg, nullptr, nullptr);
  lstm_rec<<<16, 256, 0, stream>>>(xg, Whh1p, bih[1], bhh[1], hsB);

  // Layer 2 (hsA is dead by now -> reuse)
  wmma_gemm_f16<<<dim3(4096, 4), 256, 0, stream>>>(hsB, Wih2p, 32, 512,
                                                   xg, nullptr, nullptr);
  lstm_rec<<<16, 256, 0, stream>>>(xg, Whh2p, bih[2], bhh[2], hsA);

  // FC head: pred[S,B,V] f32 = hsA @ fcW^T + fcb  (V=288 -> 18 N-tiles)
  wmma_gemm_f16<<<dim3(4096, 3), 256, 0, stream>>>(hsA, fcWp, 18, 288,
                                                   nullptr, (float*)d_out, fcb);
}
